// TSViTBlock_33638183862967
// MI455X (gfx1250) — matrix operand order repaired and checked by reading, not compiled, not run on hardware
//
#include <hip/hip_runtime.h>
#include <hip/hip_bf16.h>

#define B_ 2
#define N_ 1024
#define C_ 768
#define H_ 12
#define HP_ 24
#define HD_ 64
#define SCALE_ 0.125f

typedef __attribute__((ext_vector_type(16))) __bf16 v16bf;
typedef __attribute__((ext_vector_type(8)))  float  v8f;

struct alignas(16) US8 { unsigned short s[8]; };
union ABf { v16bf v; US8 h8[2]; };

__device__ __forceinline__ unsigned short f2bf(float x) {
    unsigned int u = __float_as_uint(x);
    u += 0x7FFFu + ((u >> 16) & 1u);            // round-to-nearest-even
    return (unsigned short)(u >> 16);
}

__device__ __forceinline__ v8f wmma_bf16(const ABf& a, const ABf& b, v8f c) {
    return __builtin_amdgcn_wmma_f32_16x16x32_bf16(
        /*neg_a=*/false, a.v, /*neg_b=*/false, b.v,
        /*c_mod=*/(short)0, c, /*reuse_a=*/false, /*reuse_b=*/false);
}

// ---------------------------------------------------------------------------
// fp32 -> bf16 conversion
// ---------------------------------------------------------------------------
__global__ void k_convert(const float* __restrict__ src,
                          unsigned short* __restrict__ dst, int n) {
    for (int i = blockIdx.x * blockDim.x + threadIdx.x; i < n;
         i += gridDim.x * blockDim.x)
        dst[i] = f2bf(src[i]);
}

// ---------------------------------------------------------------------------
// Generic tiled bf16 GEMM: Y[M,Nout] = A[M,K] * W[Nout,K]^T (+bias)
// Block = 8 waves, each wave one 16x16 tile; block tile 32x64.
// mode 0: f32 row-major (+bias); mode 1: bf16 row-major;
// mode 2: bf16 "V-swizzled" WMMA-B-ready layout (assumes M == B_*N_,
//         Nout == C_): tile index (b, m/32, h, j=d/16), within tile
//         [laneHi*16+dCol][m = 8*laneHi+0..7, 16+8*laneHi+0..7] -> one 16B store per lane.
// ---------------------------------------------------------------------------
__global__ void __launch_bounds__(256)
k_gemm(const unsigned short* __restrict__ A,
       const unsigned short* __restrict__ W,
       const float* __restrict__ bias,
       float* __restrict__ outF,
       unsigned short* __restrict__ outBF,
       int M, int Nout, int K, int mode) {
    int wave = threadIdx.x >> 5;
    int lane = threadIdx.x & 31, laneLo = lane & 15, laneHi = lane >> 4;
    int wr = wave >> 2, wc = wave & 3;
    int row0 = blockIdx.x * 32 + wr * 16;
    int col0 = blockIdx.y * 64 + wc * 16;
    if (row0 >= M || col0 >= Nout) return;   // wave-uniform

    const unsigned short* arow = A + (size_t)(row0 + laneLo) * K;
    const unsigned short* wrow = W + (size_t)(col0 + laneLo) * K;
    v8f acc = {};
    for (int k0 = 0; k0 < K; k0 += 32) {
        __builtin_prefetch(arow + k0 + 64, 0, 1);
        ABf a, b;
        // A 16x32: lane row = laneLo; two 8-elem runs
        a.h8[0] = *(const US8*)(arow + k0 + 8 * laneHi);
        a.h8[1] = *(const US8*)(arow + k0 + 16 + 8 * laneHi);
        // B 32x16: lane col = laneLo; 8-elem K runs at k0+8*laneHi, k0+16+8*laneHi
        b.h8[0] = *(const US8*)(wrow + k0 + 8 * laneHi);
        b.h8[1] = *(const US8*)(wrow + k0 + 16 + 8 * laneHi);
        acc = wmma_bf16(a, b, acc);
    }
    if (mode == 0) {
#pragma unroll
        for (int v = 0; v < 8; ++v) {
            int r = row0 + v + 8 * laneHi;
            int c = col0 + laneLo;
            outF[(size_t)r * Nout + c] = acc[v] + (bias ? bias[c] : 0.0f);
        }
    } else if (mode == 1) {
#pragma unroll
        for (int v = 0; v < 8; ++v) {
            int r = row0 + v + 8 * laneHi;
            int c = col0 + laneLo;
            outBF[(size_t)r * Nout + c] = f2bf(acc[v]);
        }
    } else {
        // acc[0..7] are 8 consecutive tokens (m) at one output column c.
        int mg = row0 + 8 * laneHi;     // first global row of the 8-run
        int bb = mg >> 10;              // / N_
        int m  = mg & (N_ - 1);
        int c  = col0 + laneLo;
        int hh = c >> 6, j = (c >> 4) & 3, dl = c & 15;
        int mp = m >> 5, mi = m & 31;   // 32-wide m-pair, index within
        US8 pk;
#pragma unroll
        for (int v = 0; v < 8; ++v) pk.s[v] = f2bf(acc[v]);
        unsigned short* dst = outBF +
            ((((size_t)(bb * 32 + mp) * H_ + hh) * 4 + j) * 32 +
             (((mi >> 3) & 1) * 16 + dl)) * 16 + (mi >> 4) * 8 + (mi & 7);
        *(US8*)dst = pk;                // single coalesced 16B store
    }
}

// ---------------------------------------------------------------------------
// Attention softmax stats: rowmax / rowsum of exp(q.k * SCALE) per (b,h,n)
// One wave per (b,h,ntile); flash-style online accumulation over m tiles.
// ---------------------------------------------------------------------------
__global__ void __launch_bounds__(256)
k_attn_stats(const unsigned short* __restrict__ QK, float2* __restrict__ stats) {
    int bh = blockIdx.x;                         // b*H_ + h
    int b = bh / H_, h = bh % H_;
    int wave = threadIdx.x >> 5;
    int ntile = blockIdx.y * 8 + wave;           // 64 ntiles
    int lane = threadIdx.x & 31, laneLo = lane & 15, laneHi = lane >> 4;
    int n0 = ntile * 16;

    const unsigned short* qb =
        QK + ((size_t)(b * N_ + n0 + laneLo)) * (2 * C_) + h * HD_;
    ABf qa[2];
#pragma unroll
    for (int t = 0; t < 2; ++t) {
        qa[t].h8[0] = *(const US8*)(qb + t * 32 + 8 * laneHi);
        qa[t].h8[1] = *(const US8*)(qb + t * 32 + 16 + 8 * laneHi);
    }
    float rmax[8], rsum[8];
#pragma unroll
    for (int v = 0; v < 8; ++v) { rmax[v] = -3.0e38f; rsum[v] = 0.f; }

    for (int m0 = 0; m0 < N_; m0 += 16) {
        const unsigned short* kb =
            QK + ((size_t)(b * N_ + m0 + laneLo)) * (2 * C_) + C_ + h * HD_;
        v8f acc = {};
#pragma unroll
        for (int t = 0; t < 2; ++t) {
            ABf kv;
            kv.h8[0] = *(const US8*)(kb + t * 32 + 8 * laneHi);
            kv.h8[1] = *(const US8*)(kb + t * 32 + 16 + 8 * laneHi);
            acc = wmma_bf16(qa[t], kv, acc);
        }
#pragma unroll
        for (int v = 0; v < 8; ++v) {
            float s  = acc[v] * SCALE_;
            float nm = fmaxf(rmax[v], s);
            rsum[v]  = rsum[v] * __expf(rmax[v] - nm) + __expf(s - nm);
            rmax[v]  = nm;
        }
    }
#pragma unroll
    for (int mask = 1; mask < 16; mask <<= 1) {
#pragma unroll
        for (int v = 0; v < 8; ++v) {
            float om = __shfl_xor(rmax[v], mask, 32);
            float os = __shfl_xor(rsum[v], mask, 32);
            float nm = fmaxf(rmax[v], om);
            rsum[v]  = rsum[v] * __expf(rmax[v] - nm) + os * __expf(om - nm);
            rmax[v]  = nm;
        }
    }
    if (laneLo == 0) {
#pragma unroll
        for (int v = 0; v < 8; ++v) {
            int n = n0 + v + 8 * laneHi;
            stats[(size_t)bh * N_ + n] = make_float2(rmax[v], rsum[v]);
        }
    }
}

// ---------------------------------------------------------------------------
// Recompute S tile per head via WMMA, softmax, expand H->HP, write A_exp.
// One wave per (b,ntile,mtile), holding all 12 heads' accumulators.
// ---------------------------------------------------------------------------
__global__ void __launch_bounds__(256)
k_attn_expand(const unsigned short* __restrict__ QK,
              const float2* __restrict__ stats,
              const float* __restrict__ Wexp, float* __restrict__ Aexp) {
    __shared__ float sW[HP_][H_];
    if (threadIdx.x < HP_ * H_)
        sW[threadIdx.x / H_][threadIdx.x % H_] = Wexp[threadIdx.x];
    __syncthreads();

    int wave = threadIdx.x >> 5;
    int lane = threadIdx.x & 31, laneLo = lane & 15, laneHi = lane >> 4;
    int b = blockIdx.x >> 6, ntile = blockIdx.x & 63;
    int mtile = blockIdx.y * 8 + wave;
    int n0 = ntile * 16, m0 = mtile * 16;

    v8f acc[H_];
#pragma unroll
    for (int h = 0; h < H_; ++h) {
        const unsigned short* qb =
            QK + ((size_t)(b * N_ + n0 + laneLo)) * (2 * C_) + h * HD_;
        const unsigned short* kb =
            QK + ((size_t)(b * N_ + m0 + laneLo)) * (2 * C_) + C_ + h * HD_;
        v8f a = {};
#pragma unroll
        for (int t = 0; t < 2; ++t) {
            ABf qa, kv;
            qa.h8[0] = *(const US8*)(qb + t * 32 + 8 * laneHi);
            qa.h8[1] = *(const US8*)(qb + t * 32 + 16 + 8 * laneHi);
            kv.h8[0] = *(const US8*)(kb + t * 32 + 8 * laneHi);
            kv.h8[1] = *(const US8*)(kb + t * 32 + 16 + 8 * laneHi);
            a = wmma_bf16(qa, kv, a);
        }
        acc[h] = a;
    }
    int m = m0 + laneLo;
#pragma unroll
    for (int v = 0; v < 8; ++v) {
        int n = n0 + v + 8 * laneHi;
        float pp[H_];
#pragma unroll
        for (int h = 0; h < H_; ++h) {
            float2 st = stats[((size_t)(b * H_ + h)) * N_ + n];
            pp[h] = __expf(acc[h][v] * SCALE_ - st.x) * (1.0f / st.y);
        }
#pragma unroll
        for (int o = 0; o < HP_; ++o) {
            float sres = 0.f;
#pragma unroll
            for (int h = 0; h < H_; ++h) sres += sW[o][h] * pp[h];
            Aexp[(((size_t)(b * HP_ + o)) * N_ + n) * N_ + m] = sres;
        }
    }
}

// ---------------------------------------------------------------------------
// Shared helper: load A_exp halo tile (24ch x 18 x 18) and produce the fused
// depthwise-3x3-conv + ReLU + head-reduce chunk sRed[16][16][H_] (* SCALE).
// Called by all 384 threads; contains block barriers.
// ---------------------------------------------------------------------------
__device__ __forceinline__ void conv_chunk(
    const float* __restrict__ Aexp, int b, int n0, int mbase, int tid,
    float (*sTile)[18][18], float (*sRed)[16][H_],
    const float (*sWloc)[9], const float* sBloc, const float (*sWred)[HP_]) {
    __syncthreads();
    for (int idx = tid; idx < HP_ * 18 * 18; idx += 384) {
        int o = idx / 324, r = idx % 324;
        int ny = r / 18, mx = r % 18;
        int n = n0 + ny - 1, m = mbase + mx - 1;
        float v = 0.f;
        if (n >= 0 && n < N_ && m >= 0 && m < N_)
            v = Aexp[(((size_t)(b * HP_ + o)) * N_ + n) * N_ + m];
        sTile[o][ny][mx] = v;
    }
    __syncthreads();
    if (tid < 256) {
        int ny = tid >> 4, mx = tid & 15;
        float ha[H_];
#pragma unroll
        for (int hh = 0; hh < H_; ++hh) ha[hh] = 0.f;
#pragma unroll
        for (int o = 0; o < HP_; ++o) {
            float c = sBloc[o];
#pragma unroll
            for (int dy = 0; dy < 3; ++dy)
#pragma unroll
                for (int dx = 0; dx < 3; ++dx)
                    c += sWloc[o][dy * 3 + dx] * sTile[o][ny + dy][mx + dx];
            c = fmaxf(c, 0.f);
#pragma unroll
            for (int hh = 0; hh < H_; ++hh) ha[hh] += sWred[hh][o] * c;
        }
#pragma unroll
        for (int hh = 0; hh < H_; ++hh) sRed[ny][mx][hh] = ha[hh] * SCALE_;
    }
    __syncthreads();
}

// ---------------------------------------------------------------------------
// Fused second stage: pass 1 computes row softmax stats (online max/sum,
// register-resident in 192 owner threads); pass 2 recomputes the conv chunk
// (mostly L2-hot), applies softmax, packs bf16 prob tiles, WMMAs against the
// swizzled V. 12 waves/block, one head per wave.
// ---------------------------------------------------------------------------
__global__ void __launch_bounds__(384)
k_apply(const float* __restrict__ Aexp, const unsigned short* __restrict__ Vswz,
        const float* __restrict__ Wloc, const float* __restrict__ bloc,
        const float* __restrict__ Wred, unsigned short* __restrict__ Ohead) {
    __shared__ float sTile[HP_][18][18];
    __shared__ float sRed[16][16][H_];
    __shared__ float2 sStats[16][H_];
    __shared__ float sWloc[HP_][9];
    __shared__ float sBloc[HP_];
    __shared__ float sWred[H_][HP_];
    int tid = threadIdx.x;
    if (tid < HP_ * 9) sWloc[tid / 9][tid % 9] = Wloc[tid];
    if (tid < HP_) sBloc[tid] = bloc[tid];
    if (tid < H_ * HP_) sWred[tid / HP_][tid % HP_] = Wred[tid];

    int b = blockIdx.x >> 6, ntile = blockIdx.x & 63;
    int n0 = ntile * 16;
    int wave = tid >> 5;                              // head id
    int lane = tid & 31, laneLo = lane & 15, laneHi = lane >> 4;
    int h = wave;

    int srow = tid / H_, sh = tid % H_;               // stats owner (tid<192)
    float rmax = -3.0e38f, rsum = 0.f;

    // ---- pass 1: second-softmax row stats for this 16-row tile
    for (int s = 0; s < 64; ++s) {
        conv_chunk(Aexp, b, n0, s * 16, tid, sTile, sRed, sWloc, sBloc, sWred);
        if (tid < 16 * H_) {
            float cmax = -3.0e38f;
            for (int m = 0; m < 16; ++m) cmax = fmaxf(cmax, sRed[srow][m][sh]);
            float nm = fmaxf(rmax, cmax);
            float add = 0.f;
            for (int m = 0; m < 16; ++m) add += __expf(sRed[srow][m][sh] - nm);
            rsum = rsum * __expf(rmax - nm) + add;
            rmax = nm;
        }
    }
    __syncthreads();
    if (tid < 16 * H_) sStats[srow][sh] = make_float2(rmax, 1.0f / rsum);
    __syncthreads();
    float rmaxv = sStats[laneLo][h].x;
    float rinv  = sStats[laneLo][h].y;

    v8f acc[4] = {{}, {}, {}, {}};
    ABf pa;
    // ---- pass 2: softmax + WMMA against swizzled V
    for (int s = 0; s < 64; ++s) {
        conv_chunk(Aexp, b, n0, s * 16, tid, sTile, sRed, sWloc, sBloc, sWred);
        int half = s & 1;
#pragma unroll
        for (int i = 0; i < 8; ++i) {
            int ml = 8 * laneHi + i;
            float p = __expf(sRed[laneLo][ml][h] - rmaxv) * rinv;
            pa.h8[half].s[i] = f2bf(p);
        }
        if (half == 1) {                              // full K=32 tile ready
            int mp = s >> 1;                          // 32-wide m-pair index
            const unsigned short* vt = Vswz +
                (((size_t)(b * 32 + mp) * H_ + h) * 4 * 32 +
                 (laneHi * 16 + laneLo)) * 16;
#pragma unroll
            for (int j = 0; j < 4; ++j) {             // HD=64 -> 4 N tiles
                ABf vb;
                vb.h8[0] = *(const US8*)(vt + (size_t)j * 512);
                vb.h8[1] = *(const US8*)(vt + (size_t)j * 512 + 8);
                acc[j] = wmma_bf16(pa, vb, acc[j]);
            }
        }
    }
#pragma unroll
    for (int j = 0; j < 4; ++j)
#pragma unroll
        for (int v = 0; v < 8; ++v) {
            int n = n0 + v + 8 * laneHi;
            int cd = h * HD_ + j * 16 + laneLo;
            Ohead[((size_t)(b * N_ + n)) * C_ + cd] = f2bf(acc[j][v]);
        }
}

// ---------------------------------------------------------------------------
extern "C" void kernel_launch(void* const* d_in, const int* in_sizes, int n_in,
                              void* d_out, int out_size, void* d_ws,
                              size_t ws_size, hipStream_t stream) {
    (void)in_sizes; (void)n_in; (void)out_size; (void)ws_size;
    const float* x     = (const float*)d_in[0];
    const float* Wqk   = (const float*)d_in[1];
    const float* Wv    = (const float*)d_in[2];
    const float* Wexp  = (const float*)d_in[3];
    const float* Wloc  = (const float*)d_in[4];
    const float* blocp = (const float*)d_in[5];
    const float* Wred  = (const float*)d_in[6];
    const float* Wproj = (const float*)d_in[7];
    const float* bproj = (const float*)d_in[8];

    float* out  = (float*)d_out;                       // (B,N,C)
    float* Aexp = out + (size_t)B_ * N_ * C_;          // (B,HP,N,N)

    char* p = (char*)d_ws;
    auto carve = [&](size_t bytes) {
        void* r = (void*)p;
        p += (bytes + 255) & ~(size_t)255;
        return r;
    };
    unsigned short* Xbf     = (unsigned short*)carve((size_t)B_ * N_ * C_ * 2);
    unsigned short* Wqkbf   = (unsigned short*)carve((size_t)2 * C_ * C_ * 2);
    unsigned short* Wvbf    = (unsigned short*)carve((size_t)C_ * C_ * 2);
    unsigned short* Wprojbf = (unsigned short*)carve((size_t)C_ * C_ * 2);
    unsigned short* QKbf    = (unsigned short*)carve((size_t)B_ * N_ * 2 * C_ * 2);
    unsigned short* Vswz    = (unsigned short*)carve((size_t)B_ * N_ * C_ * 2);
    unsigned short* Ohead   = (unsigned short*)carve((size_t)B_ * N_ * C_ * 2);
    float2* stats1 = (float2*)carve((size_t)B_ * H_ * N_ * sizeof(float2));

    dim3 blk(256);
    k_convert<<<256, blk, 0, stream>>>(x, Xbf, B_ * N_ * C_);
    k_convert<<<256, blk, 0, stream>>>(Wqk, Wqkbf, 2 * C_ * C_);
    k_convert<<<256, blk, 0, stream>>>(Wv, Wvbf, C_ * C_);
    k_convert<<<256, blk, 0, stream>>>(Wproj, Wprojbf, C_ * C_);

    // QK projection -> bf16 row-major
    k_gemm<<<dim3((B_ * N_) / 32, (2 * C_) / 64), blk, 0, stream>>>(
        Xbf, Wqkbf, nullptr, nullptr, QKbf, B_ * N_, 2 * C_, C_, 1);
    // V projection -> bf16 WMMA-B-ready swizzled layout
    k_gemm<<<dim3((B_ * N_) / 32, C_ / 64), blk, 0, stream>>>(
        Xbf, Wvbf, nullptr, nullptr, Vswz, B_ * N_, C_, C_, 2);

    k_attn_stats<<<dim3(B_ * H_, 8), blk, 0, stream>>>(QKbf, stats1);
    k_attn_expand<<<dim3(B_ * 64, 8), blk, 0, stream>>>(QKbf, stats1, Wexp,
                                                        Aexp);
    k_apply<<<dim3(B_ * 64), dim3(384), 0, stream>>>(Aexp, Vswz, Wloc, blocp,
                                                     Wred, Ohead);
    // output projection with bias -> f32 d_out
    k_gemm<<<dim3((B_ * N_) / 32, C_ / 64), blk, 0, stream>>>(
        Ohead, Wprojbf, bproj, out, nullptr, B_ * N_, C_, C_, 0);
}
